// VRNN_85804856640004
// MI455X (gfx1250) — matrix operand order repaired and checked by reading, not compiled, not run on hardware
//
#include <hip/hip_runtime.h>

typedef _Float16 half_t;
typedef __attribute__((ext_vector_type(8)))  _Float16 v8h;
typedef __attribute__((ext_vector_type(16))) _Float16 v16h;
typedef __attribute__((ext_vector_type(8)))  float    v8f;

#define SEQ_L   512
#define NBATCH  256
#define DIN     64
#define DLAT    32
#define DHID    128

// LDS layout (halves): swizzled weights, then row-major activation panels.
#define OFF_WQ   0
#define OFF_WP   (OFF_WQ  + 192*64)
#define OFF_WIH  (OFF_WP  + 160*128)
#define OFF_WHH  (OFF_WIH + 96*384)
#define OFF_PQ   (OFF_WHH + 128*384)
#define OFF_PP   (OFF_PQ  + 16*192)
#define OFF_PG   (OFF_PP  + 16*160)
#define LDS_HALVES (OFF_PG + 16*96)

// ---- CDNA5 split workgroup barrier: fence LDS stores, signal, do work, wait ----
__device__ __forceinline__ void wg_fence_signal() {
  asm volatile("s_wait_dscnt 0\n\ts_barrier_signal -1" ::: "memory");
}
__device__ __forceinline__ void wg_wait() {
  asm volatile("s_barrier_wait -1" ::: "memory");
}

// ---- Fast nonlinearities on the V_TANH_F32 / V_EXP_F32 / V_LOG_F32 trans pipes ----
__device__ __forceinline__ float fast_tanh(float x) {
#if __has_builtin(__builtin_amdgcn_tanhf)
  return __builtin_amdgcn_tanhf(x);
#else
  float r;
  asm("v_tanh_f32 %0, %1" : "=v"(r) : "v"(x));
  return r;
#endif
}
__device__ __forceinline__ float fast_sigmoid(float x) {
  return fmaf(0.5f, fast_tanh(0.5f * x), 0.5f);
}
__device__ __forceinline__ float fast_softplus(float x) {
  return (x > 20.0f) ? x : __logf(1.0f + __expf(x));
}

// Stage a (K x N) row-major f32 weight matrix into fragment-major swizzled f16 LDS:
// element j of lane l of fragment (kt,nt) sits at dst[((kt*NT+nt)*32 + l)*16 + j] and
// equals W[kt*32 + (l>>4)*16 + j][nt*16 + (l&15)]  (contiguous k order, same as load_a).
template <int K, int N>
__device__ __forceinline__ void stage_swizzled(half_t* dst, const float* __restrict__ src, int tid) {
  constexpr int NT = N / 16;
  for (int idx = tid; idx < K * N; idx += 256) {
    int j  = idx & 15;
    int l  = (idx >> 4) & 31;
    int fr = idx >> 9;
    int kt = fr / NT;
    int nt = fr - kt * NT;
    int row = kt * 32 + (l >> 4) * 16 + j;
    int col = nt * 16 + (l & 15);
    dst[idx] = (half_t)src[row * N + col];
  }
}

// One aligned 32B LDS load -> full B fragment (2x ds_load_b128, no packing VALU).
__device__ __forceinline__ v16h bfrag(const half_t* Bsw, int ntiles, int kt, int nt, int lane) {
  return *(const v16h*)(Bsw + (((kt * ntiles + nt) * 32) + lane) * 16);
}

// A fragment from a row-major panel: two aligned 16B runs (K=16*half..+7, 16*half+8..+15).
__device__ __forceinline__ v16h load_a(const half_t* A, int lda, int k0, int lane) {
  const half_t* p = A + (lane & 15) * lda + k0 + ((lane >> 4) * 16);
  v8h lo = *(const v8h*)p;
  v8h hi = *(const v8h*)(p + 8);
  return __builtin_shufflevector(lo, hi, 0, 1, 2, 3, 4, 5, 6, 7, 8, 9, 10, 11, 12, 13, 14, 15);
}

__device__ __forceinline__ v8f mm(v16h a, v16h b, v8f c) {
  return __builtin_amdgcn_wmma_f32_16x16x32_f16(false, a, false, b, (short)0, c, false, false);
}

__global__ void __launch_bounds__(256, 1)
vrnn_persistent_kernel(const float* __restrict__ X,
                       const float* __restrict__ eps_z,
                       const float* __restrict__ eps_x,
                       const float* __restrict__ Wq,  const float* __restrict__ bq,
                       const float* __restrict__ Wp,  const float* __restrict__ bp,
                       const float* __restrict__ Wih, const float* __restrict__ Whh,
                       const float* __restrict__ bih, const float* __restrict__ bhh,
                       float* __restrict__ out) {
  extern __shared__ half_t smem[];
  half_t* wqS  = smem + OFF_WQ;
  half_t* wpS  = smem + OFF_WP;
  half_t* wihS = smem + OFF_WIH;
  half_t* whhS = smem + OFF_WHH;
  half_t* pQ   = smem + OFF_PQ;   // [16][192]  cols 0..63 = x_t, 64..191 = h (f16)
  half_t* pP   = smem + OFF_PP;   // [16][160]  cols 0..31 = z,   32..159 = h (f16)
  half_t* pG   = smem + OFF_PG;   // [16][96]   cols 0..63 = x_gen, 64..95 = z (f16)

  const int tid     = threadIdx.x;
  const int wave    = tid >> 5;
  const int lane    = tid & 31;
  const int rowbase = blockIdx.x * 16;

  // One-time staging: swizzled f16 weights; zero panels (h starts at 0).
  stage_swizzled<192, 64>(wqS, Wq, tid);
  stage_swizzled<160, 128>(wpS, Wp, tid);
  stage_swizzled<96, 384>(wihS, Wih, tid);
  stage_swizzled<128, 384>(whhS, Whh, tid);
  for (int i = tid; i < 16 * 192; i += 256) pQ[i] = (half_t)0.0f;
  for (int i = tid; i < 16 * 160; i += 256) pP[i] = (half_t)0.0f;
  __syncthreads();

  // GRU weight fragments live in registers for the whole sequence (21 x v16h = 168 VGPRs).
  v16h bgi[3][3], bgh[4][3];
#pragma unroll
  for (int kt = 0; kt < 3; ++kt)
#pragma unroll
    for (int g = 0; g < 3; ++g) bgi[kt][g] = bfrag(wihS, 24, kt, wave + 8 * g, lane);
#pragma unroll
  for (int kt = 0; kt < 4; ++kt)
#pragma unroll
    for (int g = 0; g < 3; ++g) bgh[kt][g] = bfrag(whhS, 24, kt, wave + 8 * g, lane);

  // Persistent f32 hidden state: lane owns rows {i + 8*(lane>>4)} at col wave*16+(lane&15)
  float hreg[8];
#pragma unroll
  for (int i = 0; i < 8; ++i) hreg[i] = 0.0f;

  const int mrow0 = (lane >> 4) * 8;
  const int cw    = wave * 16;
  const int ncol  = lane & 15;
  const int col   = cw + ncol;

  // Hoist all loop-invariant biases into registers.
  float q_bm = 0.f, q_br = 0.f, p_bm = 0.f, p_br = 0.f;
  if (wave < 2) { q_bm = bq[col]; q_br = bq[col + 32]; }
  if (wave < 4) { p_bm = bp[col]; p_br = bp[col + 64]; }
  const float bir = bih[col], biz = bih[col + 128], bin_ = bih[col + 256];
  const float bhr = bhh[col], bhz = bhh[col + 128], bhn = bhh[col + 256];

  for (int t = 0; t < SEQ_L; ++t) {
    // ---- Phase 0: load x_t tile [16 x 64] -> pQ cols 0..63 (f16) ----
    {
      int m  = tid >> 4;
      int c0 = (tid & 15) * 4;
      const float4 xv = *(const float4*)(X + (size_t)(rowbase + m) * (SEQ_L * DIN) +
                                         (size_t)t * DIN + c0);
      half_t* dst = pQ + m * 192 + c0;
      dst[0] = (half_t)xv.x; dst[1] = (half_t)xv.y;
      dst[2] = (half_t)xv.z; dst[3] = (half_t)xv.w;
      if (t + 1 < SEQ_L)
        __builtin_prefetch(X + (size_t)(rowbase + m) * (SEQ_L * DIN) +
                           (size_t)(t + 1) * DIN + c0, 0, 0);
    }
    // Round A: pQ x-part new + h panels from previous step's writes.
    wg_fence_signal();
    wg_wait();

    // ---- Phase 1: Q head [16,192]@[192,64] + z = mean + softplus(raw)*eps_z ----
    if (wave < 2) {
      v8f cm = {}; v8f cr = {};
#pragma unroll
      for (int kt = 0; kt < 6; ++kt) {
        v16h a = load_a(pQ, 192, kt * 32, lane);
        cm = mm(a, bfrag(wqS, 4, kt, wave, lane), cm);      // mean tiles 0..1
        cr = mm(a, bfrag(wqS, 4, kt, wave + 2, lane), cr);  // raw  tiles 2..3
      }
#pragma unroll
      for (int i = 0; i < 8; ++i) {
        int m = i + mrow0;
        float mean = cm[i] + q_bm;
        float sd   = fast_softplus(cr[i] + q_br);
        float ez   = __builtin_nontemporal_load(
            eps_z + (size_t)t * (NBATCH * DLAT) + (size_t)(rowbase + m) * DLAT + col);
        half_t zh = (half_t)(mean + sd * ez);
        pP[m * 160 + col]     = zh;
        pG[m * 96 + 64 + col] = zh;
      }
    }
    // Round B: z published. gh GEMM (h @ W_hh) only needs h -> run it in the
    // barrier shadow between signal and wait on every wave.
    wg_fence_signal();
    v8f ghr = {}, ghz = {}, ghn = {};
#pragma unroll
    for (int kt = 0; kt < 4; ++kt) {
      v16h a = load_a(pQ + 64, 192, kt * 32, lane);  // A = old h (f16) inside pQ
      ghr = mm(a, bgh[kt][0], ghr);
      ghz = mm(a, bgh[kt][1], ghz);
      ghn = mm(a, bgh[kt][2], ghn);
    }
    wg_wait();

    // ---- Phase 2: P head [16,160]@[160,128] + x_gen = mean + softplus(raw)*eps_x ----
    if (wave < 4) {
      v8f cm = {}; v8f cr = {};
#pragma unroll
      for (int kt = 0; kt < 5; ++kt) {
        v16h a = load_a(pP, 160, kt * 32, lane);
        cm = mm(a, bfrag(wpS, 8, kt, wave, lane), cm);      // mean tiles 0..3
        cr = mm(a, bfrag(wpS, 8, kt, wave + 4, lane), cr);  // raw  tiles 4..7
      }
#pragma unroll
      for (int i = 0; i < 8; ++i) {
        int m = i + mrow0;
        float mean = cm[i] + p_bm;
        float sd   = fast_softplus(cr[i] + p_br);
        float ex   = __builtin_nontemporal_load(
            eps_x + (size_t)t * (NBATCH * DIN) + (size_t)(rowbase + m) * DIN + col);
        pG[m * 96 + col] = (half_t)(mean + sd * ex);
      }
    }
    // Round C: x_gen published. After this wait, ALL reads of old h are done
    // (gh consumed pQ-h before wait B, P consumed pP-h before signal C), so h
    // panels may be overwritten below without a 4th barrier.
    wg_fence_signal();
    wg_wait();

    // ---- Phase 3: GRU input GEMM + gates. Wave w owns hidden cols [16w,16w+16).
    {
      v8f gir = {}, giz = {}, gin = {};
#pragma unroll
      for (int kt = 0; kt < 3; ++kt) {
        v16h a = load_a(pG, 96, kt * 32, lane);
        gir = mm(a, bgi[kt][0], gir);
        giz = mm(a, bgi[kt][1], giz);
        gin = mm(a, bgi[kt][2], gin);
      }
#pragma unroll
      for (int i = 0; i < 8; ++i) {
        int m = i + mrow0;
        float r  = fast_sigmoid(gir[i] + bir + ghr[i] + bhr);
        float zg = fast_sigmoid(giz[i] + biz + ghz[i] + bhz);
        float nn = fast_tanh(gin[i] + bin_ + r * (ghn[i] + bhn));
        float hv = (1.0f - zg) * nn + zg * hreg[i];
        hreg[i] = hv;
        half_t hh = (half_t)hv;
        pQ[m * 192 + 64 + col] = hh;   // safe: all old-h readers passed wait C
        pP[m * 160 + 32 + col] = hh;
        __builtin_nontemporal_store(
            hv, out + (size_t)(rowbase + m) * (SEQ_L * DHID) + (size_t)t * DHID + col);
      }
      // next iteration's Round-A signal (with dscnt fence) publishes these writes
    }
  }

  // hs_final appended after output, written once from registers.
#pragma unroll
  for (int i = 0; i < 8; ++i) {
    int m = i + mrow0;
    out[(size_t)NBATCH * SEQ_L * DHID + (size_t)(rowbase + m) * DHID + col] = hreg[i];
  }
}

extern "C" void kernel_launch(void* const* d_in, const int* in_sizes, int n_in,
                              void* d_out, int out_size, void* d_ws, size_t ws_size,
                              hipStream_t stream) {
  (void)in_sizes; (void)n_in; (void)out_size; (void)d_ws; (void)ws_size;
  const float* X     = (const float*)d_in[0];
  const float* eps_z = (const float*)d_in[1];
  const float* eps_x = (const float*)d_in[2];
  const float* Wq    = (const float*)d_in[3];
  const float* bq    = (const float*)d_in[4];
  const float* Wp    = (const float*)d_in[5];
  const float* bp    = (const float*)d_in[6];
  const float* Wih   = (const float*)d_in[7];
  const float* Whh   = (const float*)d_in[8];
  const float* bih   = (const float*)d_in[9];
  const float* bhh   = (const float*)d_in[10];
  float* out = (float*)d_out;

  size_t lds_bytes = (size_t)LDS_HALVES * sizeof(_Float16);  // ~246 KB, fits 320 KB WGP LDS
  vrnn_persistent_kernel<<<dim3(NBATCH / 16), dim3(256), lds_bytes, stream>>>(
      X, eps_z, eps_x, Wq, bq, Wp, bp, Wih, Whh, bih, bhh, out);
}
